// multi_head_att_29532195127749
// MI455X (gfx1250) — matrix operand order repaired and checked by reading, not compiled, not run on hardware
//
#include <hip/hip_runtime.h>

// ---------------- problem constants ----------------
#define DMODEL 1024
#define SEQ    1024
#define NBATCH 8
#define NHEAD  8
#define DHEAD  128
#define EPS_COS 1e-8f
#define EPS_LN  1e-5f

typedef __bf16 bf16;
typedef __attribute__((ext_vector_type(16))) __bf16 v16bf;
typedef __attribute__((ext_vector_type(8)))  __bf16 v8bf;
typedef __attribute__((ext_vector_type(8)))  float  v8f;
typedef __attribute__((ext_vector_type(4)))  unsigned int u32x4;
typedef __attribute__((ext_vector_type(8)))  int i32x8;
typedef __attribute__((ext_vector_type(4)))  int i32x4;

union V16U { v16bf v; v8bf h[2]; bf16 e[16]; };

static __device__ __forceinline__ v8f v8f_zero() {
  v8f z = {0.f, 0.f, 0.f, 0.f, 0.f, 0.f, 0.f, 0.f};
  return z;
}

#define WMMA_BF16(A_, B_, C_) \
  __builtin_amdgcn_wmma_f32_16x16x32_bf16(false, (A_), false, (B_), (short)0, (C_), false, false)

#if defined(__has_builtin)
#if __has_builtin(__builtin_amdgcn_tensor_load_to_lds) && __has_builtin(__builtin_amdgcn_s_wait_tensorcnt)
#define HAVE_TDM 1
#endif
#endif

// ---------------- Tensor Data Mover: 2D bf16 tile -> LDS ----------------
// D# layout per CDNA5 ISA.
// This toolchain (clang-23 / therock) uses the 6-arg builtin:
//   (uint32x4 g0, int32x8 g1, int32x4 g2, int32x4 g3, int32x8 g4, i32 cpol)
#ifdef HAVE_TDM
static __device__ __forceinline__ void tdm_load_tile_2d_bf16(
    unsigned lds_addr, unsigned long long gaddr_bytes,
    unsigned tensor_d0, unsigned tensor_d1, unsigned stride0_elems,
    unsigned tile_d0, unsigned tile_d1) {
  u32x4 g0;
  g0[0] = 1u;                                              // count=1 (valid user D#)
  g0[1] = lds_addr;                                        // lds_addr[31:0]
  g0[2] = (unsigned)(gaddr_bytes & 0xFFFFFFFFull);         // global_addr[31:0]
  g0[3] = (unsigned)((gaddr_bytes >> 32) & 0x01FFFFFFull)  // global_addr[56:32]
          | (2u << 30);                                    // type = 2 ("image")
  i32x8 g1;
  g1[0] = (int)(1u << 16);                                 // data_size = 1 -> 2 bytes
  g1[1] = (int)((tensor_d0 & 0xFFFFu) << 16);              // tensor_dim0[15:0]
  g1[2] = (int)((tensor_d0 >> 16) | ((tensor_d1 & 0xFFFFu) << 16));
  g1[3] = (int)((tensor_d1 >> 16) | ((tile_d0 & 0xFFFFu) << 16)); // tile_dim0
  g1[4] = (int)(tile_d1 & 0xFFFFu);                        // tile_dim1 ; tile_dim2=0
  g1[5] = (int)stride0_elems;                              // tensor_dim0_stride[31:0]
  g1[6] = 0;                                               // stride0[47:32], stride1 lo
  g1[7] = 0;
  i32x4 z4 = {0, 0, 0, 0};                                 // 2D: groups 2/3 unused
  i32x8 z8 = {0, 0, 0, 0, 0, 0, 0, 0};
  __builtin_amdgcn_tensor_load_to_lds(g0, g1, z4, z4, z8, 0);
}
#endif

// ---------------- WMMA fragment loaders (contiguous k order k = 16*h + s, A and B alike) ----
// A: 16x32 bf16 tile of row-major matrix. lanes 0-15: M=lane, K {0..15};
// lanes 16-31: M=lane-16, K {16..31}. -> two contiguous 16B loads.
static __device__ __forceinline__ v16bf load_a_frag(const bf16* __restrict__ A,
                                                    int lda, int row0, int k0, int lane) {
  const int r = lane & 15, h = lane >> 4;
  const bf16* p = A + (size_t)(row0 + r) * lda + (k0 + h * 16);
  V16U u;
  u.h[0] = *(const v8bf*)p;        // K = k0 + h*16 + [0..7]
  u.h[1] = *(const v8bf*)(p + 8);  // K = k0 + h*16 + 8 + [0..7]
  return u.v;
}

// B: 32x16 bf16 tile from pre-packed weights Bp[kc][n][kk] (kk = 0..31 contiguous).
static __device__ __forceinline__ v16bf load_b_frag_packed(const bf16* __restrict__ Bp,
                                                           int N, int kc, int n0, int lane) {
  const int r = lane & 15, h = lane >> 4;
  const bf16* p = Bp + ((size_t)kc * N + (n0 + r)) * 32 + h * 16;
  V16U u;
  u.h[0] = *(const v8bf*)p;
  u.h[1] = *(const v8bf*)(p + 8);
  return u.v;
}

// B = M^T: tile B[k][n] = M[n][k] with M row-major -> contiguous along k.
static __device__ __forceinline__ v16bf load_bT_frag(const bf16* __restrict__ Mrm,
                                                     int ldm, int n0, int k0, int lane) {
  const int r = lane & 15, h = lane >> 4;
  const bf16* p = Mrm + (size_t)(n0 + r) * ldm + (k0 + h * 16);
  V16U u;
  u.h[0] = *(const v8bf*)p;
  u.h[1] = *(const v8bf*)(p + 8);
  return u.v;
}

// B from a row-major [K x N] buffer (strided gather; used on LDS-staged V tiles).
static __device__ __forceinline__ v16bf load_b_frag_rm(const bf16* __restrict__ B,
                                                       int ldb, int k0, int n0, int lane) {
  const int r = lane & 15, h = lane >> 4;
  const bf16* p = B + (size_t)(k0 + h * 16) * ldb + (n0 + r);
  V16U u;
#pragma unroll
  for (int i = 0; i < 16; ++i) u.e[i] = p[(size_t)i * ldb];
  return u.v;
}

static __device__ __forceinline__ float sigmoidf_(float x) {
  return 1.f / (1.f + __expf(-x));
}

// ---------------- elementwise / pack kernels ----------------
__global__ void __launch_bounds__(256) f2bf_kernel(const float* __restrict__ s,
                                                   bf16* __restrict__ d, long long n) {
  long long i = (long long)blockIdx.x * 256 + threadIdx.x;
  if (i < n) d[i] = (bf16)s[i];
}

// pack row-major weight W[K][N] -> Wp[kc][n][kk]
__global__ void __launch_bounds__(256) packw_kernel(const float* __restrict__ W,
                                                    bf16* __restrict__ Wp, int Kd, int N) {
  long long i = (long long)blockIdx.x * 256 + threadIdx.x;
  if (i >= (long long)Kd * N) return;
  int k = (int)(i / N), n = (int)(i % N);
  Wp[(((size_t)(k >> 5)) * N + n) * 32 + (k & 31)] = (bf16)W[(size_t)k * N + n];
}

// ---------------- generic bf16 WMMA GEMM: C = A @ Bpacked (+bias) ------------
// block = 256 threads (8 waves, 4x2), macro tile 128(M) x 64(N), 32x32 per wave.
__global__ void __launch_bounds__(256) gemm_bf16_kernel(
    const bf16* __restrict__ A, const bf16* __restrict__ Bp,
    const float* __restrict__ bias,
    bf16* __restrict__ Cbf, float* __restrict__ Cf,
    int M, int N, int K, long long sA, long long sB, long long sC) {
  const int lane = threadIdx.x & 31;
  const int wave = threadIdx.x >> 5;
  const int wm = wave >> 1, wn = wave & 1;
  const int r = lane & 15, h = lane >> 4;
  const int z = blockIdx.z;
  const int row0 = blockIdx.x * 128 + wm * 32;
  const int col0 = blockIdx.y * 64 + wn * 32;
  const bf16* Ab = A + (size_t)z * sA;
  const bf16* Bb = Bp + (size_t)z * sB;

  v8f a00 = v8f_zero(), a01 = v8f_zero(), a10 = v8f_zero(), a11 = v8f_zero();
  const int kcEnd = K >> 5;
  for (int kc = 0; kc < kcEnd; ++kc) {
    v16bf f0 = load_a_frag(Ab, K, row0, kc * 32, lane);
    v16bf f1 = load_a_frag(Ab, K, row0 + 16, kc * 32, lane);
    v16bf g0 = load_b_frag_packed(Bb, N, kc, col0, lane);
    v16bf g1 = load_b_frag_packed(Bb, N, kc, col0 + 16, lane);
    if (kc + 1 < kcEnd) {
      __builtin_prefetch(Ab + (size_t)(row0 + r) * K + (kc + 1) * 32, 0, 1);
      __builtin_prefetch(Bb + ((size_t)(kc + 1) * N + col0 + r) * 32, 0, 1);
    }
    a00 = WMMA_BF16(f0, g0, a00);
    a01 = WMMA_BF16(f0, g1, a01);
    a10 = WMMA_BF16(f1, g0, a10);
    a11 = WMMA_BF16(f1, g1, a11);
  }
  const float b0 = bias ? bias[col0 + r] : 0.f;
  const float b1 = bias ? bias[col0 + 16 + r] : 0.f;
#pragma unroll
  for (int j = 0; j < 8; ++j) {
    const int rowA = row0 + h * 8 + j;
    const int rowB = rowA + 16;
    float v00 = a00[j] + b0, v01 = a01[j] + b1;
    float v10 = a10[j] + b0, v11 = a11[j] + b1;
    if (Cbf) {
      bf16* c = Cbf + (size_t)z * sC;
      c[(size_t)rowA * N + col0 + r] = (bf16)v00;
      c[(size_t)rowA * N + col0 + 16 + r] = (bf16)v01;
      c[(size_t)rowB * N + col0 + r] = (bf16)v10;
      c[(size_t)rowB * N + col0 + 16 + r] = (bf16)v11;
    }
    if (Cf) {
      float* c = Cf + (size_t)z * sC;
      c[(size_t)rowA * N + col0 + r] = v00;
      c[(size_t)rowA * N + col0 + 16 + r] = v01;
      c[(size_t)rowB * N + col0 + r] = v10;
      c[(size_t)rowB * N + col0 + 16 + r] = v11;
    }
  }
}

// ---------------- QKV GEMM with TDM-staged A tiles -----------------
// Wave 0 DMA-loads the 128x32 bf16 A tile into LDS via tensor_load_to_lds,
// waits on TENSORcnt, then all 8 waves consume A-fragments from LDS.
__global__ void __launch_bounds__(256) gemm_bf16_tdm_kernel(
    const bf16* __restrict__ A, const bf16* __restrict__ Bp,
    const float* __restrict__ bias,
    bf16* __restrict__ Cbf, float* __restrict__ Cf,
    int M, int N, int K) {
  __shared__ bf16 ldsA[128 * 32];  // 8 KB A tile, row-major [128][32]
  const int lane = threadIdx.x & 31;
  const int wave = threadIdx.x >> 5;
  const int wm = wave >> 1, wn = wave & 1;
  const int r = lane & 15, h = lane >> 4;
  const int blockRow0 = blockIdx.x * 128;
  const int row0 = blockRow0 + wm * 32;
  const int col0 = blockIdx.y * 64 + wn * 32;

  v8f a00 = v8f_zero(), a01 = v8f_zero(), a10 = v8f_zero(), a11 = v8f_zero();
  const int kcEnd = K >> 5;
  for (int kc = 0; kc < kcEnd; ++kc) {
#ifdef HAVE_TDM
    if (wave == 0) {
      tdm_load_tile_2d_bf16(
          (unsigned)(size_t)(void*)ldsA,
          (unsigned long long)(size_t)(const void*)(A + (size_t)blockRow0 * K + kc * 32),
          (unsigned)K, (unsigned)M, (unsigned)K, 32u, 128u);
      __builtin_amdgcn_s_wait_tensorcnt(0);
    }
#else
    // fallback: cooperative copy (one 64B row segment per thread pair)
    {
      const int t = threadIdx.x;
      const bf16* src = A + (size_t)(blockRow0 + (t >> 1)) * K + kc * 32 + (t & 1) * 16;
      *(v8bf*)(ldsA + (t >> 1) * 32 + (t & 1) * 16) = *(const v8bf*)src;
      *(v8bf*)(ldsA + (t >> 1) * 32 + (t & 1) * 16 + 8) = *(const v8bf*)(src + 8);
    }
#endif
    __syncthreads();
    v16bf f0 = load_a_frag(ldsA, 32, wm * 32, 0, lane);
    v16bf f1 = load_a_frag(ldsA, 32, wm * 32 + 16, 0, lane);
    v16bf g0 = load_b_frag_packed(Bp, N, kc, col0, lane);
    v16bf g1 = load_b_frag_packed(Bp, N, kc, col0 + 16, lane);
    if (kc + 1 < kcEnd)
      __builtin_prefetch(Bp + ((size_t)(kc + 1) * N + col0 + r) * 32, 0, 1);
    a00 = WMMA_BF16(f0, g0, a00);
    a01 = WMMA_BF16(f0, g1, a01);
    a10 = WMMA_BF16(f1, g0, a10);
    a11 = WMMA_BF16(f1, g1, a11);
    __syncthreads();
  }
  const float b0 = bias ? bias[col0 + r] : 0.f;
  const float b1 = bias ? bias[col0 + 16 + r] : 0.f;
#pragma unroll
  for (int j = 0; j < 8; ++j) {
    const int rowA = row0 + h * 8 + j;
    const int rowB = rowA + 16;
    float v00 = a00[j] + b0, v01 = a01[j] + b1;
    float v10 = a10[j] + b0, v11 = a11[j] + b1;
    if (Cbf) {
      Cbf[(size_t)rowA * N + col0 + r] = (bf16)v00;
      Cbf[(size_t)rowA * N + col0 + 16 + r] = (bf16)v01;
      Cbf[(size_t)rowB * N + col0 + r] = (bf16)v10;
      Cbf[(size_t)rowB * N + col0 + 16 + r] = (bf16)v11;
    }
    if (Cf) {
      Cf[(size_t)rowA * N + col0 + r] = v00;
      Cf[(size_t)rowA * N + col0 + 16 + r] = v01;
      Cf[(size_t)rowB * N + col0 + r] = v10;
      Cf[(size_t)rowB * N + col0 + 16 + r] = v11;
    }
  }
}

// ---------------- flash attention per (batch, head, 16-row q tile) -----------
__global__ void __launch_bounds__(32) flash_attn_kernel(
    const bf16* __restrict__ Qbf, const bf16* __restrict__ Kbf,
    const bf16* __restrict__ Vbf, float* __restrict__ mh, int S) {
  __shared__ bf16 lds_p[16 * 32];
  __shared__ bf16 lds_v[32 * DHEAD];
  const int lane = threadIdx.x;
  const int r = lane & 15, h = lane >> 4;
  const int bh = blockIdx.x;
  const int batch = bh >> 3, head = bh & 7;
  const int q0 = blockIdx.y * 16;
  const bf16* Qh = Qbf + (size_t)batch * S * DMODEL + head * DHEAD;
  const bf16* Kh = Kbf + (size_t)batch * S * DMODEL + head * DHEAD;
  const bf16* Vh = Vbf + (size_t)batch * S * DMODEL + head * DHEAD;

  v16bf qf[4];
#pragma unroll
  for (int kk = 0; kk < 4; ++kk) qf[kk] = load_a_frag(Qh, DMODEL, q0, kk * 32, lane);

  v8f o[8];
#pragma unroll
  for (int t = 0; t < 8; ++t) o[t] = v8f_zero();
  float mrow[8], lrow[8];
#pragma unroll
  for (int j = 0; j < 8; ++j) { mrow[j] = -1e30f; lrow[j] = 0.f; }

  for (int kc = 0; kc < (S >> 5); ++kc) {
    // async DMA the V chunk (32 keys x 128) into LDS; overlaps with Q@K^T below
    {
      const bf16* src = Vh + (size_t)(kc * 32 + lane) * DMODEL;
      const unsigned dstl = (unsigned)(size_t)(void*)(lds_v + lane * DHEAD);
#pragma unroll
      for (int i = 0; i < DHEAD / 8; ++i) {
        asm volatile("global_load_async_to_lds_b128 %0, %1, off"
                     :: "v"(dstl + (unsigned)(i * 16)),
                        "v"((unsigned long long)(size_t)(const void*)(src + i * 8))
                     : "memory");
      }
    }
    // S = Q @ K^T for this 16x32 chunk (two 16x16 tiles, K-dim = d = 128)
    v8f s0 = v8f_zero(), s1 = v8f_zero();
#pragma unroll
    for (int kk = 0; kk < 4; ++kk) {
      v16bf g0 = load_bT_frag(Kh, DMODEL, kc * 32, kk * 32, lane);
      v16bf g1 = load_bT_frag(Kh, DMODEL, kc * 32 + 16, kk * 32, lane);
      s0 = WMMA_BF16(qf[kk], g0, s0);
      s1 = WMMA_BF16(qf[kk], g1, s1);
    }
    // online softmax; rows of C tile live across the 16 lanes of each half
#pragma unroll
    for (int j = 0; j < 8; ++j) {
      float a = s0[j] * 0.125f;   // reference divides scores by 8
      float b = s1[j] * 0.125f;
      float mx = fmaxf(a, b);
#pragma unroll
      for (int msk = 8; msk >= 1; msk >>= 1) mx = fmaxf(mx, __shfl_xor(mx, msk, 32));
      float mnew = fmaxf(mrow[j], mx);
      float corr = __expf(mrow[j] - mnew);
      float p0 = __expf(a - mnew);
      float p1 = __expf(b - mnew);
      float rs = p0 + p1;
#pragma unroll
      for (int msk = 8; msk >= 1; msk >>= 1) rs += __shfl_xor(rs, msk, 32);
      lrow[j] = lrow[j] * corr + rs;
      mrow[j] = mnew;
#pragma unroll
      for (int t = 0; t < 8; ++t) o[t][j] *= corr;
      const int prow = h * 8 + j;
      lds_p[prow * 32 + r] = (bf16)p0;
      lds_p[prow * 32 + 16 + r] = (bf16)p1;
    }
    __syncthreads();
    // complete the async V DMA before reading it back as fragments
    asm volatile("s_wait_asynccnt 0x0" ::: "memory");
    // O += P @ V (K-dim = 32 keys, 8 column tiles of 16)
    v16bf pf = load_a_frag(lds_p, 32, 0, 0, lane);
#pragma unroll
    for (int t = 0; t < 8; ++t) {
      v16bf vf = load_b_frag_rm(lds_v, DHEAD, 0, t * 16, lane);
      o[t] = WMMA_BF16(pf, vf, o[t]);
    }
    __syncthreads();
  }
  // normalize and write mh[b, q, head*128 + n] (fp32)
#pragma unroll
  for (int j = 0; j < 8; ++j) {
    float inv = 1.f / lrow[j];
    const int qrow = q0 + h * 8 + j;
    float* dst = mh + ((size_t)batch * S + qrow) * DMODEL + head * DHEAD + r;
#pragma unroll
    for (int t = 0; t < 8; ++t) dst[t * 16] = o[t][j] * inv;
  }
}

// ---------------- cosine-attention helper kernels ----------------
__global__ void __launch_bounds__(256) rownorm_kernel(
    const float* __restrict__ X, bf16* __restrict__ Xn, bf16* __restrict__ Xpack,
    int S, int Dv, float eps) {
  const int row = blockIdx.x;
  const float* x = X + (size_t)row * Dv;
  __shared__ float red[256];
  const int tid = threadIdx.x;
  float ss = 0.f;
  for (int c = tid; c < Dv; c += 256) { float v = x[c]; ss += v * v; }
  red[tid] = ss; __syncthreads();
  for (int st = 128; st >= 1; st >>= 1) {
    if (tid < st) red[tid] += red[tid + st];
    __syncthreads();
  }
  const float inv = 1.f / fmaxf(sqrtf(red[0]), eps);
  const int b_ = row / S, rs = row % S;
  for (int c = tid; c < Dv; c += 256) {
    float v = x[c];
    Xn[(size_t)row * Dv + c] = (bf16)(v * inv);
    if (Xpack) {
      size_t off = (((size_t)b_ * (S >> 5) + (rs >> 5)) * Dv + c) * 32 + (rs & 31);
      Xpack[off] = (bf16)v;
    }
  }
}

__global__ void __launch_bounds__(256) colsum_kernel(const bf16* __restrict__ kn,
                                                     float* __restrict__ Svec,
                                                     int S, int Dv) {
  const int b_ = blockIdx.y;
  const int c = blockIdx.x * 256 + threadIdx.x;
  const bf16* base = kn + (size_t)b_ * S * Dv + c;
  float s = 0.f;
  for (int k = 0; k < S; ++k) s += (float)base[(size_t)k * Dv];
  Svec[(size_t)b_ * Dv + c] = s;
}

__global__ void __launch_bounds__(256) rdot_kernel(const bf16* __restrict__ qn,
                                                   const float* __restrict__ Svec,
                                                   float* __restrict__ rvec, int S, int Dv) {
  const int row = blockIdx.x;
  const int b_ = row / S;
  const bf16* q = qn + (size_t)row * Dv;
  const float* sv = Svec + (size_t)b_ * Dv;
  __shared__ float red[256];
  const int tid = threadIdx.x;
  float acc = 0.f;
  for (int c = tid; c < Dv; c += 256) acc += (float)q[c] * sv[c];
  red[tid] = acc; __syncthreads();
  for (int st = 128; st >= 1; st >>= 1) {
    if (tid < st) red[tid] += red[tid + st];
    __syncthreads();
  }
  if (tid == 0) rvec[row] = red[0];
}

// logits L[q][k] = (qn . kn) / r[q]  (batched, B operand = kn^T contiguous)
__global__ void __launch_bounds__(256) gemm_logits_kernel(
    const bf16* __restrict__ qn, const bf16* __restrict__ kn,
    const float* __restrict__ rv, float* __restrict__ L, int S, int Dk) {
  const int lane = threadIdx.x & 31;
  const int wave = threadIdx.x >> 5;
  const int wm = wave >> 1, wn = wave & 1;
  const int r = lane & 15, h = lane >> 4;
  const int z = blockIdx.z;
  const int row0 = blockIdx.x * 128 + wm * 32;
  const int col0 = blockIdx.y * 64 + wn * 32;
  const bf16* qb = qn + (size_t)z * S * Dk;
  const bf16* kb = kn + (size_t)z * S * Dk;
  const float* rb = rv + (size_t)z * S;
  float* Lb = L + (size_t)z * S * S;

  v8f a00 = v8f_zero(), a01 = v8f_zero(), a10 = v8f_zero(), a11 = v8f_zero();
  for (int kc = 0; kc < (Dk >> 5); ++kc) {
    v16bf f0 = load_a_frag(qb, Dk, row0, kc * 32, lane);
    v16bf f1 = load_a_frag(qb, Dk, row0 + 16, kc * 32, lane);
    v16bf g0 = load_bT_frag(kb, Dk, col0, kc * 32, lane);
    v16bf g1 = load_bT_frag(kb, Dk, col0 + 16, kc * 32, lane);
    a00 = WMMA_BF16(f0, g0, a00);
    a01 = WMMA_BF16(f0, g1, a01);
    a10 = WMMA_BF16(f1, g0, a10);
    a11 = WMMA_BF16(f1, g1, a11);
  }
#pragma unroll
  for (int j = 0; j < 8; ++j) {
    const int rowA = row0 + h * 8 + j;
    const int rowB = rowA + 16;
    const float da = 1.f / rb[rowA];
    const float db = 1.f / rb[rowB];
    Lb[(size_t)rowA * S + col0 + r] = a00[j] * da;
    Lb[(size_t)rowA * S + col0 + 16 + r] = a01[j] * da;
    Lb[(size_t)rowB * S + col0 + r] = a10[j] * db;
    Lb[(size_t)rowB * S + col0 + 16 + r] = a11[j] * db;
  }
}

__global__ void __launch_bounds__(256) rowstat_kernel(const float* __restrict__ L,
                                                      float* __restrict__ mstat,
                                                      float* __restrict__ lstat, int S) {
  const int row = blockIdx.x;
  const float* x = L + (size_t)row * S;
  __shared__ float red[256];
  const int tid = threadIdx.x;
  float m = -1e30f;
  for (int c = tid; c < S; c += 256) m = fmaxf(m, x[c]);
  red[tid] = m; __syncthreads();
  for (int st = 128; st >= 1; st >>= 1) {
    if (tid < st) red[tid] = fmaxf(red[tid], red[tid + st]);
    __syncthreads();
  }
  m = red[0];
  __syncthreads();
  float s = 0.f;
  for (int c = tid; c < S; c += 256) s += __expf(x[c] - m);
  red[tid] = s; __syncthreads();
  for (int st = 128; st >= 1; st >>= 1) {
    if (tid < st) red[tid] += red[tid + st];
    __syncthreads();
  }
  if (tid == 0) { mstat[row] = m; lstat[row] = red[0]; }
}

__global__ void __launch_bounds__(256) alpha_kernel(const float* __restrict__ L,
                                                    const float* __restrict__ mstat,
                                                    const float* __restrict__ lstat,
                                                    bf16* __restrict__ alpha, int S) {
  const int row = blockIdx.y;
  const int c = blockIdx.x * 256 + threadIdx.x;
  const float m = mstat[row];
  const float inv = 1.f / lstat[row];
  const size_t idx = (size_t)row * S + c;
  alpha[idx] = (bf16)(__expf(L[idx] - m) * inv);
}

// ---------------- gated output: ao = sigmoid(cat@Wv+bv) * (cat@Wq+bq) --------
__global__ void __launch_bounds__(256) gemm_gate_kernel(
    const bf16* __restrict__ Vu, const bf16* __restrict__ Qb,
    const bf16* __restrict__ Wvp, const bf16* __restrict__ Wqp,
    const float* __restrict__ bv, const float* __restrict__ bq,
    float* __restrict__ ao, int N) {
  const int lane = threadIdx.x & 31;
  const int wave = threadIdx.x >> 5;
  const int wm = wave >> 1, wn = wave & 1;
  const int r = lane & 15, h = lane >> 4;
  const int row0 = blockIdx.x * 128 + wm * 32;
  const int col0 = blockIdx.y * 64 + wn * 32;

  v8f v00 = v8f_zero(), v01 = v8f_zero(), v10 = v8f_zero(), v11 = v8f_zero();
  v8f q00 = v8f_zero(), q01 = v8f_zero(), q10 = v8f_zero(), q11 = v8f_zero();
  for (int kc = 0; kc < 64; ++kc) {  // K = 2048: first half = V_, second = Q
    const bf16* Asrc = (kc < 32) ? Vu : Qb;
    const int k0 = (kc < 32) ? kc * 32 : (kc - 32) * 32;
    v16bf a0 = load_a_frag(Asrc, DMODEL, row0, k0, lane);
    v16bf a1 = load_a_frag(Asrc, DMODEL, row0 + 16, k0, lane);
    v16bf gv0 = load_b_frag_packed(Wvp, N, kc, col0, lane);
    v16bf gv1 = load_b_frag_packed(Wvp, N, kc, col0 + 16, lane);
    v16bf gq0 = load_b_frag_packed(Wqp, N, kc, col0, lane);
    v16bf gq1 = load_b_frag_packed(Wqp, N, kc, col0 + 16, lane);
    v00 = WMMA_BF16(a0, gv0, v00);
    v01 = WMMA_BF16(a0, gv1, v01);
    v10 = WMMA_BF16(a1, gv0, v10);
    v11 = WMMA_BF16(a1, gv1, v11);
    q00 = WMMA_BF16(a0, gq0, q00);
    q01 = WMMA_BF16(a0, gq1, q01);
    q10 = WMMA_BF16(a1, gq0, q10);
    q11 = WMMA_BF16(a1, gq1, q11);
  }
  const float bv0 = bv[col0 + r], bv1 = bv[col0 + 16 + r];
  const float bq0 = bq[col0 + r], bq1 = bq[col0 + 16 + r];
#pragma unroll
  for (int j = 0; j < 8; ++j) {
    const int rowA = row0 + h * 8 + j;
    const int rowB = rowA + 16;
    ao[(size_t)rowA * N + col0 + r] = sigmoidf_(v00[j] + bv0) * (q00[j] + bq0);
    ao[(size_t)rowA * N + col0 + 16 + r] = sigmoidf_(v01[j] + bv1) * (q01[j] + bq1);
    ao[(size_t)rowB * N + col0 + r] = sigmoidf_(v10[j] + bv0) * (q10[j] + bq0);
    ao[(size_t)rowB * N + col0 + 16 + r] = sigmoidf_(v11[j] + bv1) * (q11[j] + bq1);
  }
}

// ---------------- LayerNorm(ao + A) ----------------
__global__ void __launch_bounds__(256) layernorm_kernel(
    const float* __restrict__ ao, const float* __restrict__ res,
    const float* __restrict__ gamma, const float* __restrict__ beta,
    float* __restrict__ out, int Dv, float eps) {
  const int row = blockIdx.x;
  const float* a = ao + (size_t)row * Dv;
  const float* rs = res + (size_t)row * Dv;
  __shared__ float red1[256];
  __shared__ float red2[256];
  const int tid = threadIdx.x;
  float s1 = 0.f, s2 = 0.f;
  for (int c = tid; c < Dv; c += 256) {
    float v = a[c] + rs[c];
    s1 += v; s2 += v * v;
  }
  red1[tid] = s1; red2[tid] = s2; __syncthreads();
  for (int st = 128; st >= 1; st >>= 1) {
    if (tid < st) { red1[tid] += red1[tid + st]; red2[tid] += red2[tid + st]; }
    __syncthreads();
  }
  const float mu = red1[0] / Dv;
  const float var = red2[0] / Dv - mu * mu;
  const float inv = rsqrtf(fmaxf(var, 0.f) + eps);
  for (int c = tid; c < Dv; c += 256) {
    float v = a[c] + rs[c];
    out[(size_t)row * Dv + c] = (v - mu) * inv * gamma[c] + beta[c];
  }
}

// ---------------- host-side orchestration ----------------
extern "C" void kernel_launch(void* const* d_in, const int* in_sizes, int n_in,
                              void* d_out, int out_size, void* d_ws, size_t ws_size,
                              hipStream_t stream) {
  (void)in_sizes; (void)n_in; (void)out_size; (void)ws_size;
  const float* Bin = (const float*)d_in[0];
  const float* Ain = (const float*)d_in[1];
  const float* WQ = (const float*)d_in[2];
  const float* bQ = (const float*)d_in[3];
  const float* WK = (const float*)d_in[4];
  const float* bK = (const float*)d_in[5];
  const float* WV = (const float*)d_in[6];
  const float* bV = (const float*)d_in[7];
  const float* Wv = (const float*)d_in[8];
  const float* bv = (const float*)d_in[9];
  const float* Wq = (const float*)d_in[10];
  const float* bq = (const float*)d_in[11];
  const float* gamma = (const float*)d_in[12];
  const float* beta = (const float*)d_in[13];
  float* out = (float*)d_out;

  const size_t M = (size_t)NBATCH * SEQ;  // 8192 rows
  char* w = (char*)d_ws;
  size_t off = 0;
  auto alloc = [&](size_t bytes) -> char* {
    char* p = w + off;
    off += (bytes + 255) & ~(size_t)255;
    return p;
  };

  bf16* Bbf = (bf16*)alloc(M * DMODEL * 2);   // stage-1 A operand for Q
  bf16* Abf = (bf16*)alloc(M * DMODEL * 2);   // stage-1 A operand for K,V
  float* Lbuf = (float*)Bbf;                  // 32MB logits, reuses Bbf+Abf
  bf16* WQp = (bf16*)alloc((size_t)DMODEL * DMODEL * 2);
  bf16* WKp = (bf16*)alloc((size_t)DMODEL * DMODEL * 2);
  bf16* WVp = (bf16*)alloc((size_t)DMODEL * DMODEL * 2);
  bf16* Wvp = (bf16*)alloc((size_t)2 * DMODEL * DMODEL * 2);
  bf16* Wqp = (bf16*)alloc((size_t)2 * DMODEL * DMODEL * 2);
  bf16* Qbf = (bf16*)alloc(M * DMODEL * 2);
  bf16* Kbf = (bf16*)alloc(M * DMODEL * 2);
  bf16* alphab = Kbf;                         // reuse after flash
  bf16* Vbf = (bf16*)alloc(M * DMODEL * 2);
  bf16* Vub = Vbf;                            // reuse after flash (V_ output)
  float* Qf = (float*)alloc(M * DMODEL * 4);
  float* ao = Qf;                             // reuse after qn computed
  float* mh = (float*)alloc(M * DMODEL * 4);
  bf16* qnb = (bf16*)alloc(M * DMODEL * 2);
  bf16* knb = (bf16*)alloc(M * DMODEL * 2);
  bf16* mhp = (bf16*)alloc(M * DMODEL * 2);
  float* Svec = (float*)alloc((size_t)NBATCH * DMODEL * 4);
  float* rvec = (float*)alloc(M * 4);
  float* mstat = (float*)alloc(M * 4);
  float* lstat = (float*)alloc(M * 4);

  const long long nAct = (long long)M * DMODEL;  // 8M
  // 1) conversions + weight packing
  f2bf_kernel<<<(unsigned)(nAct / 256), 256, 0, stream>>>(Bin, Bbf, nAct);
  f2bf_kernel<<<(unsigned)(nAct / 256), 256, 0, stream>>>(Ain, Abf, nAct);
  packw_kernel<<<(unsigned)((DMODEL * DMODEL) / 256), 256, 0, stream>>>(WQ, WQp, DMODEL, DMODEL);
  packw_kernel<<<(unsigned)((DMODEL * DMODEL) / 256), 256, 0, stream>>>(WK, WKp, DMODEL, DMODEL);
  packw_kernel<<<(unsigned)((DMODEL * DMODEL) / 256), 256, 0, stream>>>(WV, WVp, DMODEL, DMODEL);
  packw_kernel<<<(unsigned)((2 * DMODEL * DMODEL) / 256), 256, 0, stream>>>(Wv, Wvp, 2 * DMODEL, DMODEL);
  packw_kernel<<<(unsigned)((2 * DMODEL * DMODEL) / 256), 256, 0, stream>>>(Wq, Wqp, 2 * DMODEL, DMODEL);

  // 2) QKV projections via TDM-staged GEMM (Q also kept fp32 for cosine stage)
  dim3 gGemm((unsigned)(M / 128), DMODEL / 64, 1);
  gemm_bf16_tdm_kernel<<<gGemm, 256, 0, stream>>>(Bbf, WQp, bQ, Qbf, Qf, (int)M, DMODEL, DMODEL);
  gemm_bf16_tdm_kernel<<<gGemm, 256, 0, stream>>>(Abf, WKp, bK, Kbf, nullptr, (int)M, DMODEL, DMODEL);
  gemm_bf16_tdm_kernel<<<gGemm, 256, 0, stream>>>(Abf, WVp, bV, Vbf, nullptr, (int)M, DMODEL, DMODEL);

  // 3) multi-head flash attention -> mh (fp32)
  flash_attn_kernel<<<dim3(NBATCH * NHEAD, SEQ / 16), 32, 0, stream>>>(Qbf, Kbf, Vbf, mh, SEQ);

  // 4) cosine-attention prep
  rownorm_kernel<<<(unsigned)M, 256, 0, stream>>>(Qf, qnb, nullptr, SEQ, DMODEL, EPS_COS);
  rownorm_kernel<<<(unsigned)M, 256, 0, stream>>>(mh, knb, mhp, SEQ, DMODEL, EPS_COS);
  colsum_kernel<<<dim3(DMODEL / 256, NBATCH), 256, 0, stream>>>(knb, Svec, SEQ, DMODEL);
  rdot_kernel<<<(unsigned)M, 256, 0, stream>>>(qnb, Svec, rvec, SEQ, DMODEL);

  // 5) logits = (qn@kn^T)/r, softmax stats, alpha
  gemm_logits_kernel<<<dim3(SEQ / 128, SEQ / 64, NBATCH), 256, 0, stream>>>(qnb, knb, rvec, Lbuf, SEQ, DMODEL);
  rowstat_kernel<<<(unsigned)M, 256, 0, stream>>>(Lbuf, mstat, lstat, SEQ);
  alpha_kernel<<<dim3(SEQ / 256, (unsigned)M), 256, 0, stream>>>(Lbuf, mstat, lstat, alphab, SEQ);

  // 6) V_ = alpha @ mh  (batched bf16 GEMM against K-packed mh)
  gemm_bf16_kernel<<<dim3(SEQ / 128, DMODEL / 64, NBATCH), 256, 0, stream>>>(
      alphab, mhp, nullptr, Vub, nullptr, SEQ, DMODEL, SEQ,
      (long long)SEQ * SEQ, (long long)SEQ * DMODEL, (long long)SEQ * DMODEL);

  // 7) gated output + 8) LayerNorm
  gemm_gate_kernel<<<dim3((unsigned)(M / 128), DMODEL / 64), 256, 0, stream>>>(
      Vub, Qbf, Wvp, Wqp, bv, bq, ao, DMODEL);
  layernorm_kernel<<<(unsigned)M, 256, 0, stream>>>(ao, Ain, gamma, beta, out, DMODEL, EPS_LN);
}
